// MultiHeadAttention_77120432767767
// MI455X (gfx1250) — matrix operand order repaired and checked by reading, compiled, not run on hardware
//
#include <hip/hip_runtime.h>
#include <hip/hip_bf16.h>

typedef __attribute__((ext_vector_type(16))) __bf16 v16bf;
typedef __attribute__((ext_vector_type(8)))  __bf16 v8bf;
typedef __attribute__((ext_vector_type(8)))  float   v8f;
typedef __attribute__((ext_vector_type(4))) unsigned int u32x4;
typedef __attribute__((ext_vector_type(8))) int          i32x8;
typedef __attribute__((ext_vector_type(4))) int          i32x4;

#define N_HEAD   8
#define D_MODEL  512
#define D_K      64
#define SEQ      1024
#define BS       2
#define MAXN     5
#define ROWS     (BS * SEQ)        // 2048
#define BH       (BS * N_HEAD)     // 16

// ---------------------------------------------------------------------------
// WMMA tile loaders (wave32; A and B both use contiguous k = 16*half + s)
// ---------------------------------------------------------------------------

// A-matrix 16x32 bf16 tile from row-major [*, lda] bf16 matrix (global or LDS).
__device__ __forceinline__ v16bf load_a_bf16(const __bf16* __restrict__ A,
                                             int lda, int m0, int k0) {
  int lane = threadIdx.x & 31;
  int half = lane >> 4;
  const __bf16* p = A + (size_t)(m0 + (lane & 15)) * lda + k0;
  v8bf lo = *(const v8bf*)(p + half * 16);
  v8bf hi = *(const v8bf*)(p + half * 16 + 8);
  v16bf r;
#pragma unroll
  for (int e = 0; e < 8; ++e) { r[e] = lo[e]; r[8 + e] = hi[e]; }
  return r;
}

// Same A tile but source is f32 (converted to bf16 on the fly).
__device__ __forceinline__ v16bf load_a_f32(const float* __restrict__ A,
                                            int lda, int m0, int k0) {
  int lane = threadIdx.x & 31;
  int half = lane >> 4;
  const float* p = A + (size_t)(m0 + (lane & 15)) * lda + k0;
  v16bf r;
#pragma unroll
  for (int e = 0; e < 8; ++e) {
    r[e]     = (__bf16)p[half * 16 + e];
    r[8 + e] = (__bf16)p[half * 16 + 8 + e];
  }
  return r;
}

// B-matrix 32x16 bf16 tile, element (kk, n) = Bt[n][kk], Bt row-major [*, ldb].
__device__ __forceinline__ v16bf load_b_bf16(const __bf16* __restrict__ Bt,
                                             int ldb, int n0, int k0) {
  int lane = threadIdx.x & 31;
  int half = lane >> 4;
  const __bf16* p = Bt + (size_t)(n0 + (lane & 15)) * ldb + k0 + half * 16;
  return *(const v16bf*)p;   // 32-byte aligned
}

#define WMMA_BF16(a, b, c) \
  __builtin_amdgcn_wmma_f32_16x16x32_bf16(false, (a), false, (b), (short)0, (c), false, false)

// ---------------------------------------------------------------------------
// Tensor Data Mover: 2D tile copy global -> LDS (CDNA5 ISA)
// D# group0: count=1 | lds_addr | 57-bit global addr | type=2 ("image")
// D# group1: data_size=2B, tensor dims / tile dims / 48-bit strides
// groups 2/3 (+ trailing group for the 6-arg builtin): zero (2D tile).
// Tracked with TENSORcnt.
// ---------------------------------------------------------------------------
__device__ __forceinline__ unsigned lds_off_of(const void* p) {
  // LDS aperture occupies addr[63:32]; addr[31:0] is the LDS byte offset.
  return (unsigned)(unsigned long long)p;
}

__device__ __forceinline__ void tdm_load_2d(unsigned lds_byte_off,
                                            const void* gptr,
                                            unsigned tile_w_elems,
                                            unsigned tile_h_rows,
                                            unsigned row_stride_elems,
                                            unsigned tensor_rows) {
  unsigned long long ga = (unsigned long long)gptr;
  unsigned long long s0 = row_stride_elems;                               // dim0 stride
  unsigned long long s1 = (unsigned long long)row_stride_elems * tensor_rows; // dim1 stride
  unsigned td0 = row_stride_elems;                                        // tensor_dim0
  unsigned td1 = tensor_rows;                                             // tensor_dim1
  u32x4 g0;
  g0[0] = 1u;                                              // count=1, user descriptor
  g0[1] = lds_byte_off;                                    // lds_addr
  g0[2] = (unsigned)(ga & 0xffffffffu);                    // global_addr[31:0]
  g0[3] = (unsigned)((ga >> 32) & 0x1ffffffu) | (2u << 30);// global_addr[56:32] | type=2
  i32x8 g1;
  g1[0] = (int)(1u << 16);                                 // data_size = 2 bytes
  g1[1] = (int)((td0 & 0xffffu) << 16);                    // tensor_dim0[15:0]
  g1[2] = (int)((td0 >> 16) | ((td1 & 0xffffu) << 16));    // td0[31:16] | td1[15:0]
  g1[3] = (int)((td1 >> 16) | (tile_w_elems << 16));       // td1[31:16] | tile_dim0
  g1[4] = (int)(tile_h_rows & 0xffffu);                    // tile_dim1 (tile_dim2 = 0)
  g1[5] = (int)(s0 & 0xffffffffu);                         // stride0[31:0]
  g1[6] = (int)(((s0 >> 32) & 0xffffu) | ((s1 & 0xffffu) << 16)); // s0[47:32]|s1[15:0]
  g1[7] = (int)((s1 >> 16) & 0xffffffffu);                 // stride1[47:16]
  i32x4 z4 = {0, 0, 0, 0};
  i32x8 z8 = {0, 0, 0, 0, 0, 0, 0, 0};
  // amdgpu-toolchain (clang-23) 6-arg form.
  __builtin_amdgcn_tensor_load_to_lds(g0, g1, z4, z4, z8, 0);
}

// ---------------------------------------------------------------------------
// Prep kernels
// ---------------------------------------------------------------------------
__global__ void ln_q_kernel(const float* __restrict__ q,
                            const float* __restrict__ gamma,
                            const float* __restrict__ beta,
                            __bf16* __restrict__ qn) {
  __shared__ float rs[256], rq[256];
  int row = blockIdx.x;
  int t = threadIdx.x;
  const float* p = q + (size_t)row * D_MODEL;
  float x0 = p[t], x1 = p[t + 256];
  rs[t] = x0 + x1;
  rq[t] = x0 * x0 + x1 * x1;
  __syncthreads();
  for (int s = 128; s > 0; s >>= 1) {
    if (t < s) { rs[t] += rs[t + s]; rq[t] += rq[t + s]; }
    __syncthreads();
  }
  float mu   = rs[0] * (1.0f / D_MODEL);
  float var  = rq[0] * (1.0f / D_MODEL) - mu * mu;
  float rstd = rsqrtf(var + 1e-6f);
  __bf16* o = qn + (size_t)row * D_MODEL;
  o[t]       = (__bf16)((x0 - mu) * rstd * gamma[t]       + beta[t]);
  o[t + 256] = (__bf16)((x1 - mu) * rstd * gamma[t + 256] + beta[t + 256]);
}

__global__ void cvt_f32_bf16_kernel(const float* __restrict__ src,
                                    __bf16* __restrict__ dst, int n) {
  int i = blockIdx.x * blockDim.x + threadIdx.x;
  if (i < n) dst[i] = (__bf16)src[i];
}

__global__ void transpose_w_kernel(const float* __restrict__ W,
                                   __bf16* __restrict__ Wt) {
  int tid = blockIdx.x * blockDim.x + threadIdx.x;   // tid = n*512 + kk
  if (tid >= D_MODEL * D_MODEL) return;
  int n  = tid >> 9;
  int kk = tid & 511;
  Wt[tid] = (__bf16)W[(size_t)kk * D_MODEL + n];
}

// ---------------------------------------------------------------------------
// Projection GEMM (mode 0: scaled qs head-major, 1: kh head-major,
// 2: v transposed [bh][64][seq])
// ---------------------------------------------------------------------------
__global__ void proj_gemm_kernel(const __bf16* __restrict__ A,
                                 const __bf16* __restrict__ Bt,
                                 __bf16* __restrict__ dst,
                                 int mode, float scale) {
  int wave = blockIdx.x * (blockDim.x >> 5) + (threadIdx.x >> 5);
  if (wave >= (ROWS / 16) * (D_MODEL / 16)) return;
  int nt = wave & 31;
  int mt = wave >> 5;
  int m0 = mt * 16, n0 = nt * 16;
  v8f acc = {};
#pragma unroll 4
  for (int k0 = 0; k0 < D_MODEL; k0 += 32) {
    v16bf a = load_a_bf16(A, D_MODEL, m0, k0);
    v16bf b = load_b_bf16(Bt, D_MODEL, n0, k0);
    acc = WMMA_BF16(a, b, acc);
  }
  int lane = threadIdx.x & 31;
  int half = lane >> 4;
  int n = n0 + (lane & 15);
  int h = n >> 6, d = n & 63;
#pragma unroll
  for (int r = 0; r < 8; ++r) {
    int m = m0 + r + half * 8;
    int bb = m >> 10, i = m & 1023;
    float vv = acc[r] * scale;
    if (mode == 2)
      dst[(((size_t)(bb * N_HEAD + h)) * D_K + d) * SEQ + i] = (__bf16)vv;
    else
      dst[(((size_t)(bb * N_HEAD + h)) * SEQ + i) * D_K + d] = (__bf16)vv;
  }
}

// ---------------------------------------------------------------------------
// qb[bh][i][m] = sum_d qs[bh][i][d] * base_rpr[m][d]   (stride padded to 8)
// ---------------------------------------------------------------------------
__global__ void qb_kernel(const __bf16* __restrict__ qs,
                          const float* __restrict__ base_rpr,
                          float* __restrict__ qb) {
  int tid = blockIdx.x * blockDim.x + threadIdx.x;
  if (tid >= BH * SEQ * 8) return;
  int m = tid & 7;
  int row = tid >> 3;
  float s = 0.0f;
  if (m <= MAXN) {
    const __bf16* qp = qs + (size_t)row * D_K;
    const float*  bp = base_rpr + m * D_K;
#pragma unroll
    for (int d = 0; d < D_K; ++d) s += (float)qp[d] * bp[d];
  }
  qb[tid] = s;
}

// ---------------------------------------------------------------------------
// Scores with TDM staging. One block (8 waves) per (bh, i-tile).
// Q tile (16x64) loaded once into LDS by TDM; K streamed in 128-row chunks,
// double-buffered, 8 waves each computing one 16x16 j-tile per chunk.
// Raw masked scores -> attn region of d_out.
// ---------------------------------------------------------------------------
__global__ void scores_tdm_kernel(const __bf16* __restrict__ qs,
                                  const __bf16* __restrict__ kh,
                                  const float* __restrict__ qb,
                                  const int* __restrict__ mask,
                                  const int* __restrict__ dist,
                                  float* __restrict__ attn) {
  __shared__ __bf16 sQ[16 * D_K];            // 2 KB
  __shared__ __bf16 sK[2][128 * D_K];        // 2 x 16 KB
  int bh = blockIdx.x >> 6;
  int it = blockIdx.x & 63;
  const __bf16* Qg = qs + ((size_t)bh * SEQ + it * 16) * D_K;
  const __bf16* Kg = kh + (size_t)bh * SEQ * D_K;

  if (threadIdx.x == 0) {
    tdm_load_2d(lds_off_of(sQ),    Qg, D_K, 16,  D_K, SEQ);
    tdm_load_2d(lds_off_of(sK[0]), Kg, D_K, 128, D_K, SEQ);
  }

  int w    = threadIdx.x >> 5;
  int lane = threadIdx.x & 31;
  int half = lane >> 4;
  int m0   = it * 16;
  int bb   = bh >> 3;
  const int* dp = dist + (size_t)bb * SEQ * SEQ;
  const int* mp = mask + (size_t)bb * SEQ * SEQ;

  for (int c = 0; c < 8; ++c) {
    if (threadIdx.x == 0) {
      if (c + 1 < 8) {
        tdm_load_2d(lds_off_of(sK[(c + 1) & 1]), Kg + (size_t)(c + 1) * 128 * D_K,
                    D_K, 128, D_K, SEQ);
        __builtin_amdgcn_s_wait_tensorcnt(1);   // chunk c (and Q) complete
      } else {
        __builtin_amdgcn_s_wait_tensorcnt(0);
      }
    }
    __syncthreads();

    const __bf16* kbuf = sK[c & 1];
    v8f acc = {};
#pragma unroll
    for (int k0 = 0; k0 < D_K; k0 += 32) {
      v16bf a = load_a_bf16(sQ, D_K, 0, k0);
      v16bf b = load_b_bf16(kbuf, D_K, w * 16, k0);
      acc = WMMA_BF16(a, b, acc);
    }
    int j = c * 128 + w * 16 + (lane & 15);
#pragma unroll
    for (int r = 0; r < 8; ++r) {
      int i = m0 + r + half * 8;
      int dc = dp[(size_t)i * SEQ + j];
      dc = dc > MAXN ? MAXN : dc;
      float s = acc[r] + qb[((size_t)bh * SEQ + i) * 8 + dc];
      if (mp[(size_t)i * SEQ + j] == 0) s = -1e9f;
      attn[((size_t)bh * SEQ + i) * SEQ + j] = s;
    }
    __syncthreads();   // protect buffer (c+1)&1 before next-iteration TDM issue
  }
}

// ---------------------------------------------------------------------------
// Softmax in place, one 1024-wide row per block (this IS the attn output).
// ---------------------------------------------------------------------------
__global__ void softmax_kernel(float* __restrict__ attn) {
  __shared__ float red[256];
  int row = blockIdx.x;
  int t = threadIdx.x;
  float4* p = (float4*)(attn + (size_t)row * SEQ);
  float4 x = p[t];
  red[t] = fmaxf(fmaxf(x.x, x.y), fmaxf(x.z, x.w));
  __syncthreads();
  for (int s = 128; s > 0; s >>= 1) {
    if (t < s) red[t] = fmaxf(red[t], red[t + s]);
    __syncthreads();
  }
  float rmax = red[0];
  __syncthreads();
  x.x = __expf(x.x - rmax); x.y = __expf(x.y - rmax);
  x.z = __expf(x.z - rmax); x.w = __expf(x.w - rmax);
  red[t] = x.x + x.y + x.z + x.w;
  __syncthreads();
  for (int s = 128; s > 0; s >>= 1) {
    if (t < s) red[t] += red[t + s];
    __syncthreads();
  }
  float inv = 1.0f / red[0];
  x.x *= inv; x.y *= inv; x.z *= inv; x.w *= inv;
  p[t] = x;
}

// ---------------------------------------------------------------------------
// O = P @ V with TDM staging of V. One block (8 waves) per (bh, 8 i-tiles).
// V^T [64 x 1024] streamed in 128-column chunks, double-buffered in LDS;
// each wave owns one i-tile and accumulates all 4 d-tiles (A reused 4x).
// ---------------------------------------------------------------------------
__global__ void pv_tdm_kernel(const float* __restrict__ attn,
                              const __bf16* __restrict__ vt,
                              __bf16* __restrict__ ob) {
  __shared__ __bf16 sV[2][D_K * 128];        // 2 x 16 KB  (rows d, 128 cols)
  int bh = blockIdx.x >> 3;
  int ig = blockIdx.x & 7;
  const float*  P  = attn + (size_t)bh * SEQ * SEQ;
  const __bf16* Vg = vt   + (size_t)bh * D_K * SEQ;

  if (threadIdx.x == 0) {
    tdm_load_2d(lds_off_of(sV[0]), Vg, 128, D_K, SEQ, D_K);
  }

  int w    = threadIdx.x >> 5;
  int lane = threadIdx.x & 31;
  int half = lane >> 4;
  int it   = ig * 8 + w;
  int m0   = it * 16;

  v8f acc0 = {}, acc1 = {}, acc2 = {}, acc3 = {};
  for (int c = 0; c < 8; ++c) {
    if (threadIdx.x == 0) {
      if (c + 1 < 8) {
        tdm_load_2d(lds_off_of(sV[(c + 1) & 1]), Vg + (size_t)(c + 1) * 128,
                    128, D_K, SEQ, D_K);
        __builtin_amdgcn_s_wait_tensorcnt(1);
      } else {
        __builtin_amdgcn_s_wait_tensorcnt(0);
      }
    }
    __syncthreads();

    const __bf16* vbuf = sV[c & 1];
#pragma unroll
    for (int k0 = 0; k0 < 128; k0 += 32) {
      v16bf a = load_a_f32(P, SEQ, m0, c * 128 + k0);   // probs -> bf16
      v16bf b0 = load_b_bf16(vbuf, 128, 0,  k0);
      v16bf b1 = load_b_bf16(vbuf, 128, 16, k0);
      v16bf b2 = load_b_bf16(vbuf, 128, 32, k0);
      v16bf b3 = load_b_bf16(vbuf, 128, 48, k0);
      acc0 = WMMA_BF16(a, b0, acc0);
      acc1 = WMMA_BF16(a, b1, acc1);
      acc2 = WMMA_BF16(a, b2, acc2);
      acc3 = WMMA_BF16(a, b3, acc3);
    }
    __syncthreads();
  }

  int h = bh & 7, bb = bh >> 3;
  int dbase = h * D_K + (lane & 15);
#pragma unroll
  for (int r = 0; r < 8; ++r) {
    size_t row = ((size_t)bb * SEQ + m0 + r + half * 8) * (N_HEAD * D_K);
    ob[row + dbase +  0] = (__bf16)acc0[r];
    ob[row + dbase + 16] = (__bf16)acc1[r];
    ob[row + dbase + 32] = (__bf16)acc2[r];
    ob[row + dbase + 48] = (__bf16)acc3[r];
  }
}

// ---------------------------------------------------------------------------
// out = O @ w_fc + residual(q)   [2048,512] f32
// ---------------------------------------------------------------------------
__global__ void fc_gemm_kernel(const __bf16* __restrict__ A,
                               const __bf16* __restrict__ Bt,
                               const float* __restrict__ residual,
                               float* __restrict__ out) {
  int wave = blockIdx.x * (blockDim.x >> 5) + (threadIdx.x >> 5);
  if (wave >= (ROWS / 16) * (D_MODEL / 16)) return;
  int nt = wave & 31;
  int mt = wave >> 5;
  int m0 = mt * 16, n0 = nt * 16;
  v8f acc = {};
#pragma unroll 4
  for (int k0 = 0; k0 < D_MODEL; k0 += 32) {
    v16bf a = load_a_bf16(A, D_MODEL, m0, k0);
    v16bf b = load_b_bf16(Bt, D_MODEL, n0, k0);
    acc = WMMA_BF16(a, b, acc);
  }
  int lane = threadIdx.x & 31;
  int half = lane >> 4;
  int n = n0 + (lane & 15);
#pragma unroll
  for (int r = 0; r < 8; ++r) {
    size_t m = (size_t)(m0 + r + half * 8);
    out[m * D_MODEL + n] = acc[r] + residual[m * D_MODEL + n];
  }
}

// ---------------------------------------------------------------------------
// Host-side orchestration
// ---------------------------------------------------------------------------
extern "C" void kernel_launch(void* const* d_in, const int* in_sizes, int n_in,
                              void* d_out, int out_size, void* d_ws, size_t ws_size,
                              hipStream_t stream) {
  const float* q        = (const float*)d_in[0];
  const float* k        = (const float*)d_in[1];
  const float* v        = (const float*)d_in[2];
  const int*   mask     = (const int*)d_in[3];
  const int*   dist     = (const int*)d_in[4];
  const float* w_qs     = (const float*)d_in[5];
  const float* w_ks     = (const float*)d_in[6];
  const float* w_vs     = (const float*)d_in[7];
  const float* w_fc     = (const float*)d_in[8];
  const float* ln_gamma = (const float*)d_in[9];
  const float* ln_beta  = (const float*)d_in[10];
  const float* base_rpr = (const float*)d_in[11];

  float* out_p  = (float*)d_out;                              // [2,1024,512]
  float* attn_p = out_p + (size_t)BS * SEQ * D_MODEL;         // [2,8,1024,1024]

  char* ws = (char*)d_ws;
  size_t off = 0;
  auto carve = [&](size_t bytes) -> char* {
    char* p = ws + off;
    off += (bytes + 255) & ~(size_t)255;
    return p;
  };
  __bf16* qn    = (__bf16*)carve((size_t)ROWS * D_MODEL * 2);
  __bf16* kb    = (__bf16*)carve((size_t)ROWS * D_MODEL * 2);
  __bf16* vb    = (__bf16*)carve((size_t)ROWS * D_MODEL * 2);
  __bf16* wqs_t = (__bf16*)carve((size_t)D_MODEL * D_MODEL * 2);
  __bf16* wks_t = (__bf16*)carve((size_t)D_MODEL * D_MODEL * 2);
  __bf16* wvs_t = (__bf16*)carve((size_t)D_MODEL * D_MODEL * 2);
  __bf16* wfc_t = (__bf16*)carve((size_t)D_MODEL * D_MODEL * 2);
  __bf16* qs    = (__bf16*)carve((size_t)BH * SEQ * D_K * 2);
  __bf16* kh    = (__bf16*)carve((size_t)BH * SEQ * D_K * 2);
  __bf16* vt    = (__bf16*)carve((size_t)BH * D_K * SEQ * 2);
  float*  qbuf  = (float*)carve((size_t)BH * SEQ * 8 * 4);
  __bf16* ob    = (__bf16*)carve((size_t)ROWS * D_MODEL * 2);

  const int nelem = ROWS * D_MODEL;   // 1048576

  // 1. prep
  ln_q_kernel<<<ROWS, 256, 0, stream>>>(q, ln_gamma, ln_beta, qn);
  cvt_f32_bf16_kernel<<<(nelem + 255) / 256, 256, 0, stream>>>(k, kb, nelem);
  cvt_f32_bf16_kernel<<<(nelem + 255) / 256, 256, 0, stream>>>(v, vb, nelem);
  transpose_w_kernel<<<1024, 256, 0, stream>>>(w_qs, wqs_t);
  transpose_w_kernel<<<1024, 256, 0, stream>>>(w_ks, wks_t);
  transpose_w_kernel<<<1024, 256, 0, stream>>>(w_vs, wvs_t);
  transpose_w_kernel<<<1024, 256, 0, stream>>>(w_fc, wfc_t);

  // 2. projections
  proj_gemm_kernel<<<1024, 128, 0, stream>>>(qn, wqs_t, qs, 0, 0.125f);
  proj_gemm_kernel<<<1024, 128, 0, stream>>>(kb, wks_t, kh, 1, 1.0f);
  proj_gemm_kernel<<<1024, 128, 0, stream>>>(vb, wvs_t, vt, 2, 1.0f);

  // 3. relative-position dots
  qb_kernel<<<(BH * SEQ * 8 + 255) / 256, 256, 0, stream>>>(qs, base_rpr, qbuf);

  // 4. scores via TDM-staged K (16 bh x 64 i-tiles blocks, 8 waves each)
  scores_tdm_kernel<<<BH * 64, 256, 0, stream>>>(qs, kh, qbuf, mask, dist, attn_p);

  // 5. softmax in place
  softmax_kernel<<<BH * SEQ, 256, 0, stream>>>(attn_p);

  // 6. P @ V via TDM-staged V (16 bh x 8 i-groups blocks)
  pv_tdm_kernel<<<BH * 8, 256, 0, stream>>>(attn_p, vt, ob);

  // 7. final projection + residual
  fc_gemm_kernel<<<1024, 128, 0, stream>>>(ob, wfc_t, q, out_p);
}
